// TFMultiheadAttention_2259152798190
// MI455X (gfx1250) — matrix operand order repaired and checked by reading, not compiled, not run on hardware
//
#include <hip/hip_runtime.h>

typedef __attribute__((ext_vector_type(16))) _Float16 v16h;
typedef __attribute__((ext_vector_type(8)))  float    v8f;

#define T_SEQ 2048
#define NB    4
#define E     768
#define NH    12
#define HD    64
#define MROWS (T_SEQ * NB)

union FragH { v16h h; uint4 q[2]; };
union H4   { uint2 u; _Float16 h[4]; };

__device__ __forceinline__ v8f wmma_f16(v16h a, v16h b, v8f c) {
  return __builtin_amdgcn_wmma_f32_16x16x32_f16(false, a, false, b, (short)0, c,
                                                false, false);
}

// ---- 16-lane xor butterfly reductions: VALU permlane16 (no LDS traffic) ----
#if defined(__has_builtin)
#if __has_builtin(__builtin_amdgcn_permlane16)
#define USE_PERMLANE16 1
#endif
#endif

#ifdef USE_PERMLANE16
__device__ __forceinline__ float pl16(float v, unsigned s0, unsigned s1) {
  int iv = __builtin_bit_cast(int, v);
  int r  = __builtin_amdgcn_permlane16(iv, iv, s0, s1, false, false);
  return __builtin_bit_cast(float, r);
}
__device__ __forceinline__ float rmax16(float v) {
  v = fmaxf(v, pl16(v, 0x67452301u, 0xEFCDAB89u));   // xor 1
  v = fmaxf(v, pl16(v, 0x54761032u, 0xDCFE98BAu));   // xor 2
  v = fmaxf(v, pl16(v, 0x32107654u, 0xBA98FEDCu));   // xor 4
  v = fmaxf(v, pl16(v, 0xFEDCBA98u, 0x76543210u));   // xor 8
  return v;
}
__device__ __forceinline__ float rsum16(float v) {
  v += pl16(v, 0x67452301u, 0xEFCDAB89u);
  v += pl16(v, 0x54761032u, 0xDCFE98BAu);
  v += pl16(v, 0x32107654u, 0xBA98FEDCu);
  v += pl16(v, 0xFEDCBA98u, 0x76543210u);
  return v;
}
#else
__device__ __forceinline__ float rmax16(float v) {
  #pragma unroll
  for (int m = 1; m < 16; m <<= 1) v = fmaxf(v, __shfl_xor(v, m, 32));
  return v;
}
__device__ __forceinline__ float rsum16(float v) {
  #pragma unroll
  for (int m = 1; m < 16; m <<= 1) v += __shfl_xor(v, m, 32);
  return v;
}
#endif

// ---- CDNA5 async global->LDS copy (ASYNCcnt-tracked LDS DMA) ----
__device__ __forceinline__ void async_b128(_Float16* lds_p, const _Float16* g) {
  // Generic LDS pointer: low 32 bits are the wave-relative LDS byte offset.
  unsigned off = (unsigned)(size_t)(void*)lds_p;
  asm volatile("global_load_async_to_lds_b128 %0, %1, off"
               :: "v"(off), "v"(g) : "memory");
}
__device__ __forceinline__ void wait_async0() {
  asm volatile("s_wait_asynccnt 0x0" ::: "memory");
}

// ---------------------------------------------------------------------------
// Tiled GEMM:  Out[M x E] = A[M x E] @ W[E x E] + bias, scaled.
// OUT_MODE: 0 = f16 row-major, 1 = f32 row-major, 2 = f16 per-head transposed
//           V layout [n][h][d][T] (feeds attention PV B-fragments directly).
// ---------------------------------------------------------------------------
template <bool A_F16, int OUT_MODE>
__global__ __launch_bounds__(128) void gemm_e(const void* __restrict__ Ap,
                                              const float* __restrict__ W,
                                              const float* __restrict__ bias,
                                              void* __restrict__ Outp,
                                              float scale) {
  __shared__ __align__(16) _Float16 Xs[64][40];
  __shared__ __align__(16) _Float16 Wt[64][40];   // Wt[n][k] = W[k][n]

  const int tid  = threadIdx.x;
  const int lane = tid & 31;
  const int w    = tid >> 5;
  const int l16  = lane & 15;
  const int bm   = blockIdx.x * 64;
  const int bn   = blockIdx.y * 64;
  const int om   = (w & 1) * 32;
  const int on   = (w >> 1) * 32;

  v8f acc[2][2] = {};

  for (int k0 = 0; k0 < E; k0 += 32) {
    #pragma unroll
    for (int i = 0; i < 4; ++i) {
      int id = tid + 128 * i;
      int r  = id >> 3;
      int c4 = (id & 7) * 4;
      if constexpr (A_F16) {
        const _Float16* A = (const _Float16*)Ap;
        *(uint2*)&Xs[r][c4] = *(const uint2*)(A + (size_t)(bm + r) * E + k0 + c4);
      } else {
        const float* A = (const float*)Ap;
        float4 v = *(const float4*)(A + (size_t)(bm + r) * E + k0 + c4);
        H4 hv;
        hv.h[0] = (_Float16)v.x; hv.h[1] = (_Float16)v.y;
        hv.h[2] = (_Float16)v.z; hv.h[3] = (_Float16)v.w;
        *(uint2*)&Xs[r][c4] = hv.u;
      }
    }
    #pragma unroll
    for (int i = 0; i < 4; ++i) {
      int id = tid + 128 * i;
      int k  = id >> 4;
      int n4 = (id & 15) * 4;
      float4 v = *(const float4*)(W + (size_t)(k0 + k) * E + bn + n4);
      Wt[n4 + 0][k] = (_Float16)v.x;
      Wt[n4 + 1][k] = (_Float16)v.y;
      Wt[n4 + 2][k] = (_Float16)v.z;
      Wt[n4 + 3][k] = (_Float16)v.w;
    }
    __syncthreads();

    const int ka = (lane < 16) ? 0 : 8;
    const int kb = (lane < 16) ? 0 : 8;
    FragH a[2], b[2];
    #pragma unroll
    for (int mi = 0; mi < 2; ++mi) {
      int m = om + mi * 16 + l16;
      a[mi].q[0] = *(const uint4*)&Xs[m][ka];
      a[mi].q[1] = *(const uint4*)&Xs[m][ka + 16];
    }
    #pragma unroll
    for (int ni = 0; ni < 2; ++ni) {
      int n = on + ni * 16 + l16;
      b[ni].q[0] = *(const uint4*)&Wt[n][kb];
      b[ni].q[1] = *(const uint4*)&Wt[n][kb + 16];
    }
    #pragma unroll
    for (int mi = 0; mi < 2; ++mi)
      #pragma unroll
      for (int ni = 0; ni < 2; ++ni)
        acc[mi][ni] = wmma_f16(a[mi].h, b[ni].h, acc[mi][ni]);
    __syncthreads();
  }

  const int mh = (lane < 16) ? 0 : 8;
  #pragma unroll
  for (int ni = 0; ni < 2; ++ni) {
    int gc = bn + on + ni * 16 + l16;
    float bv = bias[gc];
    #pragma unroll
    for (int mi = 0; mi < 2; ++mi) {
      #pragma unroll
      for (int r = 0; r < 8; ++r) {
        int gr = bm + om + mi * 16 + r + mh;
        float v = (acc[mi][ni][r] + bv) * scale;
        if constexpr (OUT_MODE == 1) {
          ((float*)Outp)[(size_t)gr * E + gc] = v;
        } else if constexpr (OUT_MODE == 2) {
          int t = gr >> 2, nb = gr & 3, hh = gc >> 6, dd = gc & 63;
          ((_Float16*)Outp)[(((size_t)nb * NH + hh) * HD + dd) * T_SEQ + t] =
              (_Float16)v;
        } else {
          ((_Float16*)Outp)[(size_t)gr * E + gc] = (_Float16)v;
        }
      }
    }
  }
}

// ---------------------------------------------------------------------------
// Flash attention: block = 64 query rows x one (batch, head).
// 4 waves x 16 query rows; 64-key tiles double-buffered in LDS via
// global_load_async_to_lds_b128; Vt is already [d][T] per head.
// ---------------------------------------------------------------------------
__global__ __launch_bounds__(128) void flash_attn(const _Float16* __restrict__ Qh,
                                                  const _Float16* __restrict__ Kh,
                                                  const _Float16* __restrict__ Vt,
                                                  _Float16* __restrict__ Oh) {
  __shared__ __align__(16) _Float16 Ks[2][64][72];    // K tile [s][d]
  __shared__ __align__(16) _Float16 Vst[2][64][72];   // V tile [d][s]
  __shared__ __align__(16) _Float16 Ps[4][16][72];    // per-wave P scratch

  const int tid  = threadIdx.x;
  const int lane = tid & 31;
  const int w    = tid >> 5;
  const int l16  = lane & 15;
  const int n    = blockIdx.z;
  const int h    = blockIdx.y;
  const int qm   = blockIdx.x * 64 + w * 16;
  const int hcol = h * HD;
  const size_t vbase = ((size_t)n * NH + h) * HD * T_SEQ;

  // resident Q fragments (2 d-slabs of 32)
  FragH qf[2];
  {
    const _Float16* qrow = Qh + (size_t)((qm + l16) * NB + n) * E + hcol;
    const int ka = (lane < 16) ? 0 : 8;
    #pragma unroll
    for (int c = 0; c < 2; ++c) {
      qf[c].q[0] = *(const uint4*)(qrow + c * 32 + ka);
      qf[c].q[1] = *(const uint4*)(qrow + c * 32 + ka + 16);
    }
  }

  float mrun[8], lrun[8];
  v8f acco[4] = {};
  #pragma unroll
  for (int r = 0; r < 8; ++r) { mrun[r] = -1e30f; lrun[r] = 0.f; }

  const int kb = (lane < 16) ? 0 : 8;
  const int ka = (lane < 16) ? 0 : 8;
  const int mh = (lane < 16) ? 0 : 8;

  const int row  = tid >> 1;          // 0..63
  const int half = (tid & 1) * 32;    // 0 or 32

  auto stage = [&](int s0, int buf) {
    const _Float16* gk = Kh + (size_t)((s0 + row) * NB + n) * E + hcol + half;
    const _Float16* gv = Vt + vbase + (size_t)row * T_SEQ + s0 + half;
    #pragma unroll
    for (int j = 0; j < 4; ++j) {
      async_b128(&Ks[buf][row][half + j * 8],  gk + j * 8);
      async_b128(&Vst[buf][row][half + j * 8], gv + j * 8);
    }
  };

  stage(0, 0);
  const int NTILE = T_SEQ / 64;
  for (int it = 0; it < NTILE; ++it) {
    const int buf = it & 1;
    wait_async0();
    __syncthreads();                       // tile `it` resident for all waves
    if (it + 1 < NTILE) stage((it + 1) * 64, buf ^ 1);   // overlap next DMA

    // ---- logits: 4 s-subtiles x 2 d-slabs = 8 WMMAs ----
    v8f accL[4] = {};
    #pragma unroll
    for (int si = 0; si < 4; ++si) {
      int sl = si * 16 + l16;
      #pragma unroll
      for (int c = 0; c < 2; ++c) {
        FragH b;
        b.q[0] = *(const uint4*)&Ks[buf][sl][c * 32 + kb];
        b.q[1] = *(const uint4*)&Ks[buf][sl][c * 32 + kb + 16];
        accL[si] = wmma_f16(qf[c].h, b.h, accL[si]);
      }
    }

    // ---- online softmax over 64 keys ----
    float mnew[8], alpha[8];
    #pragma unroll
    for (int r = 0; r < 8; ++r) {
      float mv = fmaxf(fmaxf(accL[0][r], accL[1][r]),
                       fmaxf(accL[2][r], accL[3][r]));
      mv       = rmax16(mv);
      mnew[r]  = fmaxf(mrun[r], mv);
      alpha[r] = __expf(mrun[r] - mnew[r]);
      lrun[r] *= alpha[r];
      mrun[r]  = mnew[r];
    }
    #pragma unroll
    for (int dt = 0; dt < 4; ++dt)
      #pragma unroll
      for (int r = 0; r < 8; ++r) acco[dt][r] *= alpha[r];

    #pragma unroll
    for (int si = 0; si < 4; ++si)
      #pragma unroll
      for (int r = 0; r < 8; ++r) {
        float p = __expf(accL[si][r] - mnew[r]);
        accL[si][r] = p;
        Ps[w][r + mh][si * 16 + l16] = (_Float16)p;   // wave-private, DS in-order
      }
    #pragma unroll
    for (int r = 0; r < 8; ++r)
      lrun[r] += rsum16((accL[0][r] + accL[1][r]) + (accL[2][r] + accL[3][r]));

    // ---- O += P @ V : 4 d-tiles x 2 s-slabs = 8 WMMAs ----
    FragH pf[2];
    #pragma unroll
    for (int c = 0; c < 2; ++c) {
      pf[c].q[0] = *(const uint4*)&Ps[w][l16][c * 32 + ka];
      pf[c].q[1] = *(const uint4*)&Ps[w][l16][c * 32 + ka + 16];
    }
    #pragma unroll
    for (int dt = 0; dt < 4; ++dt) {
      int dl = dt * 16 + l16;
      #pragma unroll
      for (int c = 0; c < 2; ++c) {
        FragH b;
        b.q[0] = *(const uint4*)&Vst[buf][dl][c * 32 + kb];
        b.q[1] = *(const uint4*)&Vst[buf][dl][c * 32 + kb + 16];
        acco[dt] = wmma_f16(pf[c].h, b.h, acco[dt]);
      }
    }
  }

  // ---- normalize + store f16 attention output (row-major [token][E]) ----
  #pragma unroll
  for (int dt = 0; dt < 4; ++dt)
    #pragma unroll
    for (int r = 0; r < 8; ++r) {
      int t = qm + r + mh;
      float v = acco[dt][r] / lrun[r];
      Oh[(size_t)(t * NB + n) * E + hcol + dt * 16 + l16] = (_Float16)v;
    }
}

// ---------------------------------------------------------------------------
extern "C" void kernel_launch(void* const* d_in, const int* in_sizes, int n_in,
                              void* d_out, int out_size, void* d_ws, size_t ws_size,
                              hipStream_t stream) {
  (void)in_sizes; (void)n_in; (void)out_size; (void)ws_size;
  const float* query = (const float*)d_in[0];
  const float* key   = (const float*)d_in[1];
  const float* value = (const float*)d_in[2];
  const float* Wq = (const float*)d_in[3];
  const float* bq = (const float*)d_in[4];
  const float* Wk = (const float*)d_in[5];
  const float* bk = (const float*)d_in[6];
  const float* Wv = (const float*)d_in[7];
  const float* bv = (const float*)d_in[8];
  const float* Wo = (const float*)d_in[9];
  const float* bo = (const float*)d_in[10];
  float* out = (float*)d_out;

  const size_t ME = (size_t)MROWS * E;
  _Float16* Qh = (_Float16*)d_ws;
  _Float16* Kh = Qh + ME;
  _Float16* Vt = Kh + ME;   // per-head transposed V [n][h][d][T]
  _Float16* Oh = Vt + ME;

  dim3 gg(MROWS / 64, E / 64, 1);
  gemm_e<false, 0><<<gg, 128, 0, stream>>>(query, Wq, bq, Qh, 0.125f);
  gemm_e<false, 0><<<gg, 128, 0, stream>>>(key,   Wk, bk, Kh, 1.0f);
  gemm_e<false, 2><<<gg, 128, 0, stream>>>(value, Wv, bv, Vt, 1.0f);

  dim3 ga(T_SEQ / 64, NH, NB);
  flash_attn<<<ga, 128, 0, stream>>>(Qh, Kh, Vt, Oh);

  gemm_e<true, 1><<<gg, 128, 0, stream>>>(Oh, Wo, bo, out, 1.0f);
}
